// FEM_11029476016518
// MI455X (gfx1250) — matrix operand order repaired and checked by reading, not compiled, not run on hardware
//
#include <hip/hip_runtime.h>

#define B_  4
#define C_  96
#define H_  128
#define W_  128
#define HW_ 16384

typedef _Float16 h16v __attribute__((ext_vector_type(16)));
typedef float    f8v  __attribute__((ext_vector_type(8)));

// Load a 16-half WMMA fragment from two 16B chunks (p0 -> halves 0..7, p1 -> halves 8..15).
static __device__ __forceinline__ h16v ld_frag(const _Float16* p0, const _Float16* p1) {
    union { h16v v; uint4 q[2]; } u;
    u.q[0] = *reinterpret_cast<const uint4*>(p0);
    u.q[1] = *reinterpret_cast<const uint4*>(p1);
    return u.v;
}
static __device__ __forceinline__ f8v wmma16(h16v a, h16v b, f8v c) {
    return __builtin_amdgcn_wmma_f32_16x16x32_f16(false, a, false, b, (short)0, c, false, false);
}
static __device__ __forceinline__ f8v fzero() {
    f8v z = {0.f,0.f,0.f,0.f,0.f,0.f,0.f,0.f};
    return z;
}

// ---------------------------------------------------------------- zero energy
__global__ void k_zero(float* energy) {
    int i = blockIdx.x * 256 + threadIdx.x;
    if (i < B_ * HW_) energy[i] = 0.f;
}

// -------------------------------------------------- conv weight B-fragments
// K order: input(2) x slot(9) x cin(96)  -> K=1728, 54 k-steps. N = cout = 96 (6 tiles).
// frag[((nt*54+s)*32+lane)*16+j] = Weff[cout=nt*16+(lane&15)][k = s*32+((j&8)<<1)+((lane&16)>>1)+(j&7)]
__global__ void k_frag_conv(const float* w1, const float* lm1, const float* th1,
                            const float* w2, const float* lm2, const float* th2,
                            _Float16* out) {
    int gid = blockIdx.x * 256 + threadIdx.x;
    const int total = 6 * 54 * 512;
    if (gid >= total) return;
    int j    = gid & 15;
    int lane = (gid >> 4) & 31;
    int s    = (gid >> 9) % 54;
    int nt   = gid / (512 * 54);
    int cout = nt * 16 + (lane & 15);
    int k    = s * 32 + ((j & 8) << 1) + ((lane & 16) >> 1) + (j & 7);
    int inp  = k / 864;
    int r    = k - inp * 864;
    int slot = r / 96;
    int cin  = r - slot * 96;
    const float* w  = inp ? w2  : w1;
    const float* lm = inp ? lm2 : lm1;
    float theta     = (inp ? th2 : th1)[0];
    int base = (cout * 96 + cin) * 9;
    float val = w[base + slot];
    if (slot == 4) {
        float ssum = 0.f;
        #pragma unroll
        for (int q = 0; q < 9; ++q) ssum += w[base + q];
        float delta = theta * lm[cout * 96 + cin] * ssum;
        val = val - delta * val;
    }
    out[gid] = (_Float16)val;
}

// ------------------------------------------- generic linear-weight fragments
// W is (N,K) row-major; B[k][n] = W[rowoff+n][k].
__global__ void k_frag_lin(const float* Wm, int ldk, int rowoff,
                           _Float16* out, int nKs, int total) {
    int gid = blockIdx.x * 256 + threadIdx.x;
    if (gid >= total) return;
    int j    = gid & 15;
    int lane = (gid >> 4) & 31;
    int s    = (gid >> 9) % nKs;
    int nt   = gid / (512 * nKs);
    int n = nt * 16 + (lane & 15);
    int k = s * 32 + ((j & 8) << 1) + ((lane & 16) >> 1) + (j & 7);
    out[gid] = (_Float16)Wm[(size_t)(rowoff + n) * ldk + k];
}

// ------------------------------------------------ DCT matrix B-fragments
// D[n][k] = cos(pi*(2k+1)*n/256) * sqrt(2/128) * (n==0 ? 1/sqrt2 : 1), 128x128.
__global__ void k_frag_dct(_Float16* out) {
    int gid = blockIdx.x * 256 + threadIdx.x;
    const int total = 8 * 4 * 512;
    if (gid >= total) return;
    int j    = gid & 15;
    int lane = (gid >> 4) & 31;
    int s    = (gid >> 9) & 3;
    int nt   = gid / (512 * 4);
    int n = nt * 16 + (lane & 15);
    int k = s * 32 + ((j & 8) << 1) + ((lane & 16) >> 1) + (j & 7);
    float v = cosf(3.14159265358979f * (float)(2 * k + 1) * (float)n / 256.0f) * 0.125f;
    if (n == 0) v *= 0.70710678118f;
    out[gid] = (_Float16)v;
}

// ----------------------------------- NCHW f32 -> (b,hw,c) channel-last f16
__global__ void k_tocl(const float* x1, const float* x2, _Float16* o1, _Float16* o2) {
    __shared__ float tile[32][257];
    int b   = blockIdx.z >> 1;
    int inp = blockIdx.z & 1;
    int c0  = blockIdx.y * 32;
    int hw0 = blockIdx.x * 256;
    const float* x = inp ? x2 : x1;
    _Float16*    o = inp ? o2 : o1;
    int t = threadIdx.x;
    for (int it = 0; it < 32; ++it) {
        int idx = t + 256 * it;             // 32c x 256hw
        int cl = idx >> 8, hwl = idx & 255;
        tile[cl][hwl] = x[((size_t)(b * C_ + c0 + cl)) * HW_ + hw0 + hwl];
    }
    __syncthreads();
    for (int it = 0; it < 32; ++it) {
        int idx = t + 256 * it;
        int hwl = idx >> 5, cl = idx & 31;
        o[((size_t)(b * HW_ + hw0 + hwl)) * C_ + c0 + cl] = (_Float16)tile[cl][hwl];
    }
}

// ------------------------------------------------- fused dual conv + residual
// t = conv(x1,w1) + conv(x2,w2) + 2*(x1+x2), output channel-last f32.
__global__ void __launch_bounds__(256) k_conv(const _Float16* x1cl, const _Float16* x2cl,
                                              const _Float16* convF, float* tout) {
    extern __shared__ char smem[];
    _Float16* ls1 = (_Float16*)smem;
    _Float16* ls2 = ls1 + 18 * 18 * 96;
    int b = blockIdx.y;
    int y0 = (blockIdx.x >> 3) * 16, x0 = (blockIdx.x & 7) * 16;
    int t = threadIdx.x;
    for (int idx = t; idx < 18 * 18 * 96; idx += 256) {
        int c = idx % 96, pos = idx / 96;
        int xx = pos % 18, yy = pos / 18;
        int hh = y0 + yy - 1, ww = x0 + xx - 1;
        bool ok = (hh >= 0) & (hh < 128) & (ww >= 0) & (ww < 128);
        size_t g = ((size_t)(b * HW_ + hh * 128 + ww)) * C_ + c;
        ls1[idx] = ok ? x1cl[g] : (_Float16)0.f;
        ls2[idx] = ok ? x2cl[g] : (_Float16)0.f;
    }
    __syncthreads();
    int wave = t >> 5, lane = t & 31, l15 = lane & 15;
    int lo8 = (lane & 16) >> 1, hi8 = lo8;
    f8v acc[2][6];
    #pragma unroll
    for (int a = 0; a < 2; ++a)
        #pragma unroll
        for (int n = 0; n < 6; ++n) acc[a][n] = fzero();
    for (int s = 0; s < 54; ++s) {
        int inp = (s >= 27);
        int rem = s - inp * 27;
        int slot = rem / 3, c0 = (rem % 3) * 32;
        int ky = slot / 3, kx = slot % 3;
        const _Float16* ls = inp ? ls2 : ls1;
        h16v A[2];
        #pragma unroll
        for (int a = 0; a < 2; ++a) {
            int py = 2 * wave + a;
            int off = ((py + ky) * 18 + (l15 + kx)) * 96 + c0 + lo8;
            A[a] = ld_frag(ls + off, ls + off + 16);
        }
        #pragma unroll
        for (int nt = 0; nt < 6; ++nt) {
            const _Float16* bp = convF + ((size_t)(nt * 54 + s) * 32 + lane) * 16;
            h16v Bf = ld_frag(bp, bp + 8);
            acc[0][nt] = wmma16(A[0], Bf, acc[0][nt]);
            acc[1][nt] = wmma16(A[1], Bf, acc[1][nt]);
        }
    }
    #pragma unroll
    for (int a = 0; a < 2; ++a) {
        int py = 2 * wave + a;
        #pragma unroll
        for (int nt = 0; nt < 6; ++nt) {
            int c = nt * 16 + l15;
            #pragma unroll
            for (int r = 0; r < 8; ++r) {
                int px = r + hi8;
                int ctr = ((py + 1) * 18 + (px + 1)) * 96 + c;
                float fu = 2.f * ((float)ls1[ctr] + (float)ls2[ctr]);
                size_t g = ((size_t)(b * HW_ + (y0 + py) * 128 + x0 + px)) * C_ + c;
                tout[g] = acc[a][nt][r] + fu;
            }
        }
    }
}

// ---------------------------------- swin attention (win=1): t2 = t + proj(v(LN(t)))
__global__ void __launch_bounds__(256) k_attn(const float* tin, const _Float16* vF,
                                              const _Float16* pF, const float* g1,
                                              const float* b1, const float* qkvb,
                                              const float* projb, float* t2out) {
    extern __shared__ char smem[];
    float*    lt = (float*)smem;                       // 256*96 f32
    _Float16* la = (_Float16*)(smem + 256 * 96 * 4);   // 256*96 f16
    _Float16* lv = la + 256 * 96;                      // 256*96 f16
    int tok0 = blockIdx.x * 256;
    int t = threadIdx.x;
    for (int idx = t; idx < 256 * 96; idx += 256) lt[idx] = tin[(size_t)tok0 * 96 + idx];
    __syncthreads();
    {
        const float* row = lt + t * 96;
        float m = 0.f;
        for (int i = 0; i < 96; ++i) m += row[i];
        m *= (1.f / 96.f);
        float v = 0.f;
        for (int i = 0; i < 96; ++i) { float d = row[i] - m; v += d * d; }
        float rs = rsqrtf(v * (1.f / 96.f) + 1e-5f);
        for (int i = 0; i < 96; ++i)
            la[t * 96 + i] = (_Float16)((row[i] - m) * rs * g1[i] + b1[i]);
    }
    __syncthreads();
    int wave = t >> 5, lane = t & 31, l15 = lane & 15;
    int lo8 = (lane & 16) >> 1, hi8 = lo8;
    f8v acc[2][6];
    #pragma unroll
    for (int a = 0; a < 2; ++a)
        #pragma unroll
        for (int n = 0; n < 6; ++n) acc[a][n] = fzero();
    #pragma unroll
    for (int s = 0; s < 3; ++s) {
        h16v A[2];
        #pragma unroll
        for (int a = 0; a < 2; ++a) {
            int off = ((2 * wave + a) * 16 + l15) * 96 + s * 32 + lo8;
            A[a] = ld_frag(la + off, la + off + 16);
        }
        #pragma unroll
        for (int nt = 0; nt < 6; ++nt) {
            const _Float16* bp = vF + ((size_t)(nt * 3 + s) * 32 + lane) * 16;
            h16v Bf = ld_frag(bp, bp + 8);
            acc[0][nt] = wmma16(A[0], Bf, acc[0][nt]);
            acc[1][nt] = wmma16(A[1], Bf, acc[1][nt]);
        }
    }
    #pragma unroll
    for (int a = 0; a < 2; ++a)
        #pragma unroll
        for (int nt = 0; nt < 6; ++nt) {
            int col = nt * 16 + l15;
            #pragma unroll
            for (int r = 0; r < 8; ++r) {
                int rowi = (2 * wave + a) * 16 + r + hi8;
                lv[rowi * 96 + col] = (_Float16)(acc[a][nt][r] + qkvb[192 + col]);
            }
        }
    __syncthreads();
    #pragma unroll
    for (int a = 0; a < 2; ++a)
        #pragma unroll
        for (int n = 0; n < 6; ++n) acc[a][n] = fzero();
    #pragma unroll
    for (int s = 0; s < 3; ++s) {
        h16v A[2];
        #pragma unroll
        for (int a = 0; a < 2; ++a) {
            int off = ((2 * wave + a) * 16 + l15) * 96 + s * 32 + lo8;
            A[a] = ld_frag(lv + off, lv + off + 16);
        }
        #pragma unroll
        for (int nt = 0; nt < 6; ++nt) {
            const _Float16* bp = pF + ((size_t)(nt * 3 + s) * 32 + lane) * 16;
            h16v Bf = ld_frag(bp, bp + 8);
            acc[0][nt] = wmma16(A[0], Bf, acc[0][nt]);
            acc[1][nt] = wmma16(A[1], Bf, acc[1][nt]);
        }
    }
    #pragma unroll
    for (int a = 0; a < 2; ++a)
        #pragma unroll
        for (int nt = 0; nt < 6; ++nt) {
            int col = nt * 16 + l15;
            #pragma unroll
            for (int r = 0; r < 8; ++r) {
                int rowi = (2 * wave + a) * 16 + r + hi8;
                t2out[(size_t)(tok0 + rowi) * 96 + col] =
                    lt[rowi * 96 + col] + acc[a][nt][r] + projb[col];
            }
        }
}

// ------------------------------- MLP: cross = t2 + W2( gelu(W1(LN(t2))+b1) ) + b2
__global__ void __launch_bounds__(256) k_mlp(const float* t2in, const _Float16* m1F,
                                             const _Float16* m2F, const float* g2,
                                             const float* b2, const float* mb1,
                                             const float* mb2, float* crossout) {
    extern __shared__ char smem[];
    float*    lt = (float*)smem;                        // 128*96 f32
    _Float16* la = (_Float16*)(smem + 128 * 96 * 4);    // 128*96 f16
    _Float16* lh = la + 128 * 96;                       // 128*384 f16
    int tok0 = blockIdx.x * 128;
    int t = threadIdx.x;
    for (int idx = t; idx < 128 * 96; idx += 256) lt[idx] = t2in[(size_t)tok0 * 96 + idx];
    __syncthreads();
    if (t < 128) {
        const float* row = lt + t * 96;
        float m = 0.f;
        for (int i = 0; i < 96; ++i) m += row[i];
        m *= (1.f / 96.f);
        float v = 0.f;
        for (int i = 0; i < 96; ++i) { float d = row[i] - m; v += d * d; }
        float rs = rsqrtf(v * (1.f / 96.f) + 1e-5f);
        for (int i = 0; i < 96; ++i)
            la[t * 96 + i] = (_Float16)((row[i] - m) * rs * g2[i] + b2[i]);
    }
    __syncthreads();
    int wave = t >> 5, lane = t & 31, l15 = lane & 15;
    int lo8 = (lane & 16) >> 1, hi8 = lo8;
    int rowbase = wave * 16;
    h16v Af[3];
    #pragma unroll
    for (int s = 0; s < 3; ++s) {
        int off = (rowbase + l15) * 96 + s * 32 + lo8;
        Af[s] = ld_frag(la + off, la + off + 16);
    }
    for (int p = 0; p < 2; ++p) {
        f8v acc[12];
        #pragma unroll
        for (int q = 0; q < 12; ++q) acc[q] = fzero();
        #pragma unroll
        for (int s = 0; s < 3; ++s)
            #pragma unroll
            for (int q = 0; q < 12; ++q) {
                int nt = p * 12 + q;
                const _Float16* bp = m1F + ((size_t)(nt * 3 + s) * 32 + lane) * 16;
                h16v Bf = ld_frag(bp, bp + 8);
                acc[q] = wmma16(Af[s], Bf, acc[q]);
            }
        #pragma unroll
        for (int q = 0; q < 12; ++q) {
            int col = (p * 12 + q) * 16 + l15;
            #pragma unroll
            for (int r = 0; r < 8; ++r) {
                int rowi = rowbase + r + hi8;
                float x = acc[q][r] + mb1[col];
                float ge = 0.5f * x * (1.f + erff(x * 0.70710678118f));
                lh[rowi * 384 + col] = (_Float16)ge;
            }
        }
    }
    f8v acc2[6];
    #pragma unroll
    for (int n = 0; n < 6; ++n) acc2[n] = fzero();
    for (int s = 0; s < 12; ++s) {
        int off = (rowbase + l15) * 384 + s * 32 + lo8;
        h16v A = ld_frag(lh + off, lh + off + 16);
        #pragma unroll
        for (int nt = 0; nt < 6; ++nt) {
            const _Float16* bp = m2F + ((size_t)(nt * 12 + s) * 32 + lane) * 16;
            h16v Bf = ld_frag(bp, bp + 8);
            acc2[nt] = wmma16(A, Bf, acc2[nt]);
        }
    }
    #pragma unroll
    for (int nt = 0; nt < 6; ++nt) {
        int col = nt * 16 + l15;
        #pragma unroll
        for (int r = 0; r < 8; ++r) {
            int rowi = rowbase + r + hi8;
            crossout[(size_t)(tok0 + rowi) * 96 + col] =
                lt[rowi * 96 + col] + acc2[nt][r] + mb2[col];
        }
    }
}

// ------------------ cross (b,hw,c) f32 -> Xt (b, w*96+c, h) f16 (h contiguous)
__global__ void k_xt(const float* cross, _Float16* Xt) {
    __shared__ float tl[32][97];
    int b = blockIdx.z, wq = blockIdx.y, h0 = blockIdx.x * 32;
    int t = threadIdx.x;
    for (int idx = t; idx < 32 * 96; idx += 256) {
        int hh = idx / 96, c = idx % 96;
        tl[hh][c] = cross[((size_t)(b * HW_ + (h0 + hh) * 128 + wq)) * C_ + c];
    }
    __syncthreads();
    for (int idx = t; idx < 32 * 96; idx += 256) {
        int c = idx >> 5, hh = idx & 31;
        Xt[((size_t)(b * 12288 + wq * 96 + c)) * 128 + h0 + hh] = (_Float16)tl[hh][c];
    }
}

// -------- DCT pass 1: T1[(b,c,i,w)] = sum_h Xt[(b,w,c,h)] * D[i][h]
__global__ void __launch_bounds__(256) k_dct1(const _Float16* Xt, const _Float16* dctF,
                                              _Float16* T1) {
    int b = blockIdx.y, Mbase = blockIdx.x * 256;
    int t = threadIdx.x, wave = t >> 5, lane = t & 31, l15 = lane & 15;
    int lo8 = (lane & 16) >> 1, hi8 = lo8;
    for (int p = 0; p < 2; ++p) {
        f8v acc[2][4];
        #pragma unroll
        for (int a = 0; a < 2; ++a)
            #pragma unroll
            for (int q = 0; q < 4; ++q) acc[a][q] = fzero();
        #pragma unroll
        for (int s = 0; s < 4; ++s) {
            h16v A[2];
            #pragma unroll
            for (int a = 0; a < 2; ++a) {
                int m = Mbase + (2 * wave + a) * 16 + l15;
                const _Float16* ap = Xt + ((size_t)b * 12288 + m) * 128 + s * 32 + lo8;
                A[a] = ld_frag(ap, ap + 16);
            }
            #pragma unroll
            for (int q = 0; q < 4; ++q) {
                int nt = p * 4 + q;
                const _Float16* bp = dctF + ((size_t)(nt * 4 + s) * 32 + lane) * 16;
                h16v Bf = ld_frag(bp, bp + 8);
                acc[0][q] = wmma16(A[0], Bf, acc[0][q]);
                acc[1][q] = wmma16(A[1], Bf, acc[1][q]);
            }
        }
        #pragma unroll
        for (int a = 0; a < 2; ++a)
            #pragma unroll
            for (int q = 0; q < 4; ++q) {
                int i = (p * 4 + q) * 16 + l15;
                #pragma unroll
                for (int r = 0; r < 8; ++r) {
                    int m = Mbase + (2 * wave + a) * 16 + r + hi8;
                    int c = m % 96, wl = m / 96;
                    T1[((size_t)(b * 96 + c) * 128 + i) * 128 + wl] = (_Float16)acc[a][q][r];
                }
            }
    }
}

// -------- DCT pass 2: energy[b][i*128+j] += |sum_w T1[(b,c,i,w)]*D[j][w]| / 96
__global__ void __launch_bounds__(256) k_dct2(const _Float16* T1, const _Float16* dctF,
                                              float* energy) {
    int b = blockIdx.y, Mbase = blockIdx.x * 256;
    int t = threadIdx.x, wave = t >> 5, lane = t & 31, l15 = lane & 15;
    int lo8 = (lane & 16) >> 1, hi8 = lo8;
    for (int p = 0; p < 2; ++p) {
        f8v acc[2][4];
        #pragma unroll
        for (int a = 0; a < 2; ++a)
            #pragma unroll
            for (int q = 0; q < 4; ++q) acc[a][q] = fzero();
        #pragma unroll
        for (int s = 0; s < 4; ++s) {
            h16v A[2];
            #pragma unroll
            for (int a = 0; a < 2; ++a) {
                int m = Mbase + (2 * wave + a) * 16 + l15;
                const _Float16* ap = T1 + ((size_t)b * 12288 + m) * 128 + s * 32 + lo8;
                A[a] = ld_frag(ap, ap + 16);
            }
            #pragma unroll
            for (int q = 0; q < 4; ++q) {
                int nt = p * 4 + q;
                const _Float16* bp = dctF + ((size_t)(nt * 4 + s) * 32 + lane) * 16;
                h16v Bf = ld_frag(bp, bp + 8);
                acc[0][q] = wmma16(A[0], Bf, acc[0][q]);
                acc[1][q] = wmma16(A[1], Bf, acc[1][q]);
            }
        }
        #pragma unroll
        for (int a = 0; a < 2; ++a)
            #pragma unroll
            for (int q = 0; q < 4; ++q) {
                int jcol = (p * 4 + q) * 16 + l15;
                #pragma unroll
                for (int r = 0; r < 8; ++r) {
                    int m = Mbase + (2 * wave + a) * 16 + r + hi8;
                    int i = m & 127;
                    atomicAdd(energy + (size_t)b * HW_ + i * 128 + jcol,
                              fabsf(acc[a][q][r]) * (1.0f / 96.f));
                }
            }
    }
}

// ---------------- top-96 selection + tiny MLP gate -> scale[b] = 1 + sigmoid(...)
__global__ void k_gate(const float* energy, const float* fw1, const float* fb1,
                       const float* fw2, const float* fb2, float* scale) {
    extern __shared__ char smem[];
    float* se   = (float*)smem;       // 16384
    float* rv   = se + 16384;         // 256
    int*   ri   = (int*)(rv + 256);   // 256
    float* tk   = (float*)(ri + 256); // 96
    float* hbuf = tk + 96;            // 24
    int b = blockIdx.x, t = threadIdx.x;
    for (int idx = t; idx < 16384; idx += 256) se[idx] = energy[(size_t)b * HW_ + idx];
    __syncthreads();
    for (int r = 0; r < 96; ++r) {
        float bv = -1e30f; int bi = 0;
        for (int j = t; j < 16384; j += 256)
            if (se[j] > bv) { bv = se[j]; bi = j; }
        rv[t] = bv; ri[t] = bi;
        __syncthreads();
        for (int s = 128; s > 0; s >>= 1) {
            if (t < s && rv[t + s] > rv[t]) { rv[t] = rv[t + s]; ri[t] = ri[t + s]; }
            __syncthreads();
        }
        if (t == 0) { tk[r] = rv[0]; se[ri[0]] = -1e30f; }
        __syncthreads();
    }
    if (t < 24) {
        float h = fb1[t];
        for (int k = 0; k < 96; ++k) h += fw1[t * 96 + k] * tk[k];
        hbuf[t] = h > 0.f ? h : 0.f;
    }
    __syncthreads();
    if (t == 0) {
        float o = fb2[0];
        for (int j = 0; j < 24; ++j) o += fw2[j] * hbuf[j];
        scale[b] = 1.f + 1.f / (1.f + expf(-o));
    }
}

// ---------------- final: out1 = x1 + cross*scale, out2 = x2 + cross*scale (NCHW)
__global__ void k_out(const float* cross, const float* x1, const float* x2,
                      const float* scale, float* out) {
    __shared__ float tl[256][33];
    int b = blockIdx.z, c0 = blockIdx.y * 32, hw0 = blockIdx.x * 256;
    int t = threadIdx.x;
    for (int idx = t; idx < 8192; idx += 256) {
        int hw = idx >> 5, cl = idx & 31;
        tl[hw][cl] = cross[((size_t)(b * HW_ + hw0 + hw)) * C_ + c0 + cl];
    }
    __syncthreads();
    float s = scale[b];
    for (int idx = t; idx < 8192; idx += 256) {
        int cl = idx >> 8, hwl = idx & 255;
        float v = tl[hwl][cl] * s;
        size_t g = ((size_t)(b * C_ + c0 + cl)) * HW_ + hw0 + hwl;
        out[g] = x1[g] + v;
        out[(size_t)6291456 + g] = x2[g] + v;
    }
}

extern "C" void kernel_launch(void* const* d_in, const int* in_sizes, int n_in,
                              void* d_out, int out_size, void* d_ws, size_t ws_size,
                              hipStream_t stream) {
    (void)in_sizes; (void)n_in; (void)out_size; (void)ws_size;
    const float* x1    = (const float*)d_in[0];
    const float* x2    = (const float*)d_in[1];
    const float* w_tx1 = (const float*)d_in[2];
    const float* lm1   = (const float*)d_in[3];
    const float* th1   = (const float*)d_in[4];
    const float* w_tx2 = (const float*)d_in[5];
    const float* lm2   = (const float*)d_in[6];
    const float* th2   = (const float*)d_in[7];
    const float* ln1g  = (const float*)d_in[8];
    const float* ln1b  = (const float*)d_in[9];
    const float* qkvw  = (const float*)d_in[10];
    const float* qkvb  = (const float*)d_in[11];
    const float* projw = (const float*)d_in[12];
    const float* projb = (const float*)d_in[13];
    const float* ln2g  = (const float*)d_in[14];
    const float* ln2b  = (const float*)d_in[15];
    const float* mw1   = (const float*)d_in[16];
    const float* mb1   = (const float*)d_in[17];
    const float* mw2   = (const float*)d_in[18];
    const float* mb2   = (const float*)d_in[19];
    const float* fw1   = (const float*)d_in[20];
    const float* fb1   = (const float*)d_in[21];
    const float* fw2   = (const float*)d_in[22];
    const float* fb2   = (const float*)d_in[23];
    float* out = (float*)d_out;

    char* ws = (char*)d_ws;
    // workspace layout (bytes)
    _Float16* convF = (_Float16*)(ws + 0);         // 331776
    _Float16* vF    = (_Float16*)(ws + 331776);    // 18432
    _Float16* pF    = (_Float16*)(ws + 350208);    // 18432
    _Float16* m1F   = (_Float16*)(ws + 368640);    // 73728
    _Float16* m2F   = (_Float16*)(ws + 442368);    // 73728
    _Float16* dctF  = (_Float16*)(ws + 516096);    // 32768
    // RB: x1cl/x2cl (f16) during conv; cross (f32) afterwards  (25165824 B)
    _Float16* x1cl  = (_Float16*)(ws + 1048576);
    _Float16* x2cl  = x1cl + 6291456;
    float*    cross = (float*)(ws + 1048576);
    // RC: t (f32) during conv/attn; Xt + T1 (f16) during DCT   (25165824 B)
    float*    tbuf  = (float*)(ws + 26214400);
    _Float16* Xt    = (_Float16*)(ws + 26214400);
    _Float16* T1    = (_Float16*)(ws + 26214400 + 12582912);
    // RD: t2 (f32)
    float*    t2    = (float*)(ws + 51380224);
    float*    energy = (float*)(ws + 76546048);
    float*    scale  = (float*)(ws + 76808192);

    dim3 blk(256);
    // 0) zero energy accumulator (required every call)
    k_zero<<<dim3(256), blk, 0, stream>>>(energy);
    // 1) weight fragments
    k_frag_conv<<<dim3((6 * 54 * 512 + 255) / 256), blk, 0, stream>>>(
        w_tx1, lm1, th1, w_tx2, lm2, th2, convF);
    k_frag_lin<<<dim3(36), blk, 0, stream>>>(qkvw, 96, 192, vF, 3, 6 * 3 * 512);
    k_frag_lin<<<dim3(36), blk, 0, stream>>>(projw, 96, 0, pF, 3, 6 * 3 * 512);
    k_frag_lin<<<dim3(144), blk, 0, stream>>>(mw1, 96, 0, m1F, 3, 24 * 3 * 512);
    k_frag_lin<<<dim3(144), blk, 0, stream>>>(mw2, 384, 0, m2F, 12, 6 * 12 * 512);
    k_frag_dct<<<dim3(64), blk, 0, stream>>>(dctF);
    // 2) channel-last f16 copies of x1, x2
    k_tocl<<<dim3(64, 3, 8), blk, 0, stream>>>(x1, x2, x1cl, x2cl);
    // 3) fused dual LDC conv + 2*(x1+x2) -> t
    k_conv<<<dim3(64, 4), blk, 2 * 18 * 18 * 96 * 2, stream>>>(x1cl, x2cl, convF, tbuf);
    // 4) attention (win=1) -> t2
    k_attn<<<dim3(256), blk, 256 * 96 * 4 + 2 * 256 * 96 * 2, stream>>>(
        tbuf, vF, pF, ln1g, ln1b, qkvb, projb, t2);
    // 5) MLP -> cross  (overwrites x1cl/x2cl region; they are dead now)
    k_mlp<<<dim3(512), blk, 128 * 96 * 4 + 128 * 96 * 2 + 128 * 384 * 2, stream>>>(
        t2, m1F, m2F, ln2g, ln2b, mb1, mb2, cross);
    // 6) DCT: transpose, two WMMA GEMM passes, energy accumulation
    k_xt<<<dim3(4, 128, 4), blk, 0, stream>>>(cross, Xt);
    k_dct1<<<dim3(48, 4), blk, 0, stream>>>(Xt, dctF, T1);
    k_dct2<<<dim3(48, 4), blk, 0, stream>>>(T1, dctF, energy);
    // 7) top-96 + gate MLP -> scale
    k_gate<<<dim3(4), blk, (16384 + 256 + 256 + 96 + 24 + 8) * 4, stream>>>(
        energy, fw1, fb1, fw2, fb2, scale);
    // 8) final residual outputs (NCHW)
    k_out<<<dim3(64, 3, 4), blk, 0, stream>>>(cross, x1, x2, scale, out);
}
